// sphericalConv_82643760710361
// MI455X (gfx1250) — matrix operand order repaired and checked by reading, not compiled, not run on hardware
//
#include <hip/hip_runtime.h>
#include <cstdint>
#include <cstddef>

#define NPIX 196608
#define NBATCH 4
#define CIN 32
#define COUTC 64
#define KNB 9

typedef __attribute__((ext_vector_type(16))) _Float16 v16h;
typedef __attribute__((ext_vector_type(8)))  float    v8f;

union Frag16 { uint32_t u[8]; uint4 q[2]; v16h h; };
union PackH  { _Float16 f[2]; uint32_t u; };

// ---------------------------------------------------------------------------
// Pre-pass: x (B, CIN, NPIX) f32  ->  xT (B, NPIX, CIN) f16.
// Reads are lane-coalesced (consecutive lanes = consecutive pixels); each
// thread emits one 64B channel-last row as 4x b128 stores.
// ---------------------------------------------------------------------------
__global__ __launch_bounds__(256) void xpose_f16_kernel(
    const float* __restrict__ x, _Float16* __restrict__ xT)
{
    const int p = blockIdx.x * 256 + threadIdx.x;
#pragma unroll
    for (int b = 0; b < NBATCH; ++b) {
        // four bases keep channel-stride immediates < 8MB (24-bit IOFFSET)
        const float* xs0 = x + (size_t)b * CIN * NPIX + p;
        const float* xs1 = xs0 + (size_t)8  * NPIX;
        const float* xs2 = xs0 + (size_t)16 * NPIX;
        const float* xs3 = xs0 + (size_t)24 * NPIX;
        union { _Float16 f[32]; uint4 q[4]; } r;
#pragma unroll
        for (int c = 0; c < 8; ++c) {
            r.f[c +  0] = (_Float16)xs0[(size_t)c * NPIX];
            r.f[c +  8] = (_Float16)xs1[(size_t)c * NPIX];
            r.f[c + 16] = (_Float16)xs2[(size_t)c * NPIX];
            r.f[c + 24] = (_Float16)xs3[(size_t)c * NPIX];
        }
        uint4* dst = (uint4*)(xT + ((size_t)b * NPIX + p) * CIN);
#pragma unroll
        for (int i = 0; i < 4; ++i) dst[i] = r.q[i];
    }
}

// ---------------------------------------------------------------------------
// B-fragment gathers
// ---------------------------------------------------------------------------
// Channel-last f16 path: one lane's 16-channel window at neighbour np is
// 32 contiguous, 32B-aligned bytes -> two global_load_b128, already in
// this kernel's B order, k = 16h + s (dword v = channels {2v, 2v+1}), same as A.
__device__ __forceinline__ Frag16 gather_bfrag_t(
    const _Float16* __restrict__ xtb, int np, int h)
{
    Frag16 bf;
    const uint4* q = (const uint4*)(xtb + (size_t)np * CIN + h * 16);
    bf.q[0] = q[0];
    bf.q[1] = q[1];
    return bf;
}

// Fallback (no scratch): 16 strided f32 gathers + cvt, split bases for IOFFSET.
__device__ __forceinline__ Frag16 gather_bfrag(const float* __restrict__ xb, int np) {
    Frag16 bf;
    const float* xs  = xb + np;
    const float* xs8 = xs + (size_t)8 * NPIX;
#pragma unroll
    for (int v = 0; v < 4; ++v) {
        bf.h[2 * v + 0] = (_Float16)xs[(size_t)(2 * v + 0) * NPIX];
        bf.h[2 * v + 1] = (_Float16)xs[(size_t)(2 * v + 1) * NPIX];
    }
#pragma unroll
    for (int v = 4; v < 8; ++v) {
        bf.h[2 * v + 0] = (_Float16)xs8[(size_t)(2 * v - 8) * NPIX];
        bf.h[2 * v + 1] = (_Float16)xs8[(size_t)(2 * v - 7) * NPIX];
    }
    return bf;
}

// ---------------------------------------------------------------------------
// Main kernel: one wave handles TWO 16-pixel tiles (32 pixels); each LDS A
// fragment feeds two WMMAs. D = A(16x32 weights) x B(32x16 gathered pixels).
// Reduction order ck' = k*32 + c: k-step s uses one neighbour idx[p,s], the
// 32 K-locals are the input channels.
// ---------------------------------------------------------------------------
template <bool USE_XT>
__global__ __launch_bounds__(128) void spconv_wmma_kernel(
    const float* __restrict__ x,        // (B, CIN, NPIX) f32
    const _Float16* __restrict__ xT,    // (B, NPIX, CIN) f16 (USE_XT only)
    const long long* __restrict__ nbr,  // (NPIX*9) int64
    const float* __restrict__ W,        // (COUT, CIN, K)
    const float* __restrict__ bias,     // (COUT,)
    float* __restrict__ out)            // (B, COUT, NPIX)
{
    // Weight fragments in A-matrix VGPR layout: [ct(4)][s(9)][v(8)][lane(32)]
    __shared__ uint32_t wlds[4 * 9 * 8 * 32];   // 36 KB

    const int tid  = threadIdx.x;
    const int lane = tid & 31;
    const int wave = tid >> 5;
    const int m    = lane & 15;   // A-row / B-col / D-col within tile
    const int h    = lane >> 4;   // lane half

    // ---- stage f16 weight fragments into LDS (A layout, k = 16h + s order as in B) ----
    for (int i = tid; i < 4 * 9 * 8 * 32; i += 128) {
        const int l  = i & 31;
        const int v  = (i >> 5) & 7;
        const int cs = i >> 8;          // ct*9 + s
        const int s  = cs % 9;
        const int ct = cs / 9;
        const int mm = l & 15;
        const int hh = l >> 4;
        const int o  = ct * 16 + mm;
        const int c0 = hh * 16 + v * 2;
        // A[m][k_local] = W[o][c = k_local][kk = s]
        const float w0 = W[(size_t)o * (CIN * KNB) + (c0 + 0) * KNB + s];
        const float w1 = W[(size_t)o * (CIN * KNB) + (c0 + 1) * KNB + s];
        PackH pk;
        pk.f[0] = (_Float16)w0;
        pk.f[1] = (_Float16)w1;
        wlds[i] = pk.u;
    }
    __syncthreads();

    const int wt = blockIdx.x * 4 + wave;   // 32-pixel super-tile
    const int p0 = wt * 32 + m;
    const int p1 = p0 + 16;

    int np0[KNB], np1[KNB];
#pragma unroll
    for (int k = 0; k < KNB; ++k) {
        np0[k] = (int)nbr[(size_t)p0 * KNB + k];
        np1[k] = (int)nbr[(size_t)p1 * KNB + k];
    }

    const float* xl = x + (size_t)(h * 16) * NPIX;   // fallback path base

    float bv[4][8];
#pragma unroll
    for (int ct = 0; ct < 4; ++ct)
#pragma unroll
        for (int v = 0; v < 8; ++v)
            bv[ct][v] = bias[ct * 16 + h * 8 + v];

    for (int b = 0; b < NBATCH; ++b) {
        const float*    xb  = xl + (size_t)b * CIN * NPIX;
        const _Float16* xtb = xT + (size_t)b * NPIX * CIN;

        v8f acc0[4] = { {}, {}, {}, {} };
        v8f acc1[4] = { {}, {}, {}, {} };

#pragma unroll
        for (int s = 0; s < KNB; ++s) {
            Frag16 bf0, bf1;
            if constexpr (USE_XT) {
                bf0 = gather_bfrag_t(xtb, np0[s], h);
                bf1 = gather_bfrag_t(xtb, np1[s], h);
            } else {
                bf0 = gather_bfrag(xb, np0[s]);
                bf1 = gather_bfrag(xb, np1[s]);
            }
#pragma unroll
            for (int ct = 0; ct < 4; ++ct) {
                Frag16 af;
#pragma unroll
                for (int v = 0; v < 8; ++v)
                    af.u[v] = wlds[((ct * 9 + s) * 8 + v) * 32 + lane];
                acc0[ct] = __builtin_amdgcn_wmma_f32_16x16x32_f16(
                    false, af.h, false, bf0.h, (short)0, acc0[ct], false, false);
                acc1[ct] = __builtin_amdgcn_wmma_f32_16x16x32_f16(
                    false, af.h, false, bf1.h, (short)0, acc1[ct], false, false);
            }
        }

        // ---- store D tiles: lane(n,h) VGPR v -> row o = ct*16 + 8h + v ----
#pragma unroll
        for (int ct = 0; ct < 4; ++ct) {
            const int o = ct * 16 + h * 8;
            const size_t rowbase = ((size_t)b * COUTC + o) * NPIX;
#pragma unroll
            for (int v = 0; v < 8; ++v) {
                out[rowbase + (size_t)v * NPIX + (size_t)wt * 32 + m] =
                    acc0[ct][v] + bv[ct][v];
                out[rowbase + (size_t)v * NPIX + (size_t)wt * 32 + 16 + m] =
                    acc1[ct][v] + bv[ct][v];
            }
        }
    }
}

extern "C" void kernel_launch(void* const* d_in, const int* in_sizes, int n_in,
                              void* d_out, int out_size, void* d_ws, size_t ws_size,
                              hipStream_t stream) {
    (void)in_sizes; (void)n_in; (void)out_size;
    const float*     x    = (const float*)d_in[0];
    const long long* nbr  = (const long long*)d_in[1];
    const float*     W    = (const float*)d_in[2];
    const float*     bias = (const float*)d_in[3];
    float*           out  = (float*)d_out;

    const int supertiles = NPIX / 32;        // 6144 thirty-two-pixel super-tiles
    const int blocks     = supertiles / 4;   // 4 waves (wave32) per 128-thread block

    const size_t xt_bytes = (size_t)NBATCH * NPIX * CIN * sizeof(_Float16); // 48 MB
    if (ws_size >= xt_bytes) {
        _Float16* xT = (_Float16*)d_ws;
        xpose_f16_kernel<<<NPIX / 256, 256, 0, stream>>>(x, xT);
        spconv_wmma_kernel<true><<<blocks, 128, 0, stream>>>(x, xT, nbr, W, bias, out);
    } else {
        spconv_wmma_kernel<false><<<blocks, 128, 0, stream>>>(x, nullptr, nbr, W, bias, out);
    }
}
